// FixedPointSolver_72696616452469
// MI455X (gfx1250) — matrix operand order repaired and checked by reading, not compiled, not run on hardware
//
#include <hip/hip_runtime.h>
#include <stdint.h>

#define D_DIM 128
#define P_DIM 128
#define N_DIM 8192
#define TOLF  1e-5f
#define MAX_ITER 50
#define NB 32                 // columns of A per workgroup -> 256 workgroups
#define KB 32                 // K per WMMA step (bf16 16x16x32)
#define NSTEPS (N_DIM / KB)   // 256 (multiple of 4)
#define LDS_STRIDE 40         // bf16 elems per transposed column (16B-aligned frags)

typedef __attribute__((ext_vector_type(16))) __bf16 v16bf;
typedef __attribute__((ext_vector_type(8)))  __bf16 v8bf;
typedef __attribute__((ext_vector_type(8)))  float  v8f;

struct SolverState {
    unsigned err_bits;   // atomic-max accumulator (float bits, values >= 0)
    float    prev_err;
    float    alpha;
    int      converged;
};

__device__ __forceinline__ unsigned short f2bf(float f) {
    union { float f; unsigned u; } v; v.f = f;
    unsigned u = v.u;
    unsigned r = u + 0x7fffu + ((u >> 16) & 1u);   // round-to-nearest-even
    return (unsigned short)(r >> 16);
}
// pack bf16(lo) | bf16(hi)<<16 : 2x RNE add + one v_perm_b32 byte select
__device__ __forceinline__ unsigned pack2bf(float lo, float hi) {
    unsigned ulo = __float_as_uint(lo);
    unsigned uhi = __float_as_uint(hi);
    ulo += 0x7fffu + ((ulo >> 16) & 1u);
    uhi += 0x7fffu + ((uhi >> 16) & 1u);
    return __builtin_amdgcn_perm(uhi, ulo, 0x07060302u);  // {uhi[31:16], ulo[31:16]}
}

// LDS-only workgroup barrier: orders DS ops without draining global LOADcnt,
// so the global prefetch pipeline survives across K-steps.
__device__ __forceinline__ void wg_barrier_lds() {
    asm volatile("s_wait_dscnt 0x0\n\t"
                 "s_barrier_signal -1\n\t"
                 "s_barrier_wait -1" ::: "memory");
}

__device__ __forceinline__ v16bf load_afrag(const unsigned short* tp) {
    v8bf lo = *(const v8bf*)tp;
    v8bf hi = *(const v8bf*)(tp + 16);
    return __builtin_shufflevector(lo, hi, 0,1,2,3,4,5,6,7,8,9,10,11,12,13,14,15);
}
__device__ __forceinline__ v16bf load_bfrag(const unsigned short* bp) {
    v8bf lo = *(const v8bf*)bp;
    v8bf hi = *(const v8bf*)(bp + 16);
    return __builtin_shufflevector(lo, hi, 0,1,2,3,4,5,6,7,8,9,10,11,12,13,14,15);
}
// every thread stages 2 rows x 2 cols of the 32x32 A tile
__device__ __forceinline__ void load_tile(const float* __restrict__ Ag, int ks, int n0,
                                          int r_ld, int c_ld, float2& a0, float2& a1) {
    const float* s0 = Ag + (size_t)(ks * KB + r_ld) * N_DIM + (n0 + c_ld);
    a0 = *(const float2*)s0;
    a1 = *(const float2*)(s0 + N_DIM);
}
__device__ __forceinline__ void store_tile(unsigned short* buf, int c_ld, int r_ld,
                                           float2 a0, float2 a1) {
    *(unsigned*)&buf[(c_ld    ) * LDS_STRIDE + r_ld] = pack2bf(a0.x, a1.x);
    *(unsigned*)&buf[(c_ld + 1) * LDS_STRIDE + r_ld] = pack2bf(a0.y, a1.y);
}

// out[d][n] = sum_p L[d][p] * R[p][n]   (L: 128x128 f32, R: 128x8192 f32)
// d = blockIdx.y -> L row accesses are wave-uniform (scalar loads).
__global__ __launch_bounds__(256)
void small_gemm_kernel(const float* __restrict__ L, const float* __restrict__ R,
                       float* __restrict__ outF, unsigned short* __restrict__ outB,
                       const SolverState* __restrict__ st, int guarded) {
    if (guarded && st->converged) return;
    const int d = blockIdx.y;
    const int n = blockIdx.x * 256 + threadIdx.x;
    const float* Lr = L + d * P_DIM;       // uniform across the block
    const float* Rc = R + n;
    float acc = 0.f;
#pragma unroll 8
    for (int p = 0; p < P_DIM; ++p)
        acc = fmaf(Lr[p], Rc[(size_t)p * N_DIM], acc);
    const size_t idx = (size_t)d * N_DIM + n;
    if (outF) outF[idx] = acc;
    if (outB) outB[idx] = f2bf(acc);
}

__global__ void init_kernel(float* __restrict__ Z, SolverState* st) {
    int idx = blockIdx.x * blockDim.x + threadIdx.x;
    if (idx < D_DIM * N_DIM) Z[idx] = 0.f;
    if (idx == 0) { st->err_bits = 0u; st->prev_err = 1.0f; st->alpha = 0.5f; st->converged = 0; }
}

__global__ void commit_kernel(float* __restrict__ Z, const float* __restrict__ ZN,
                              const SolverState* __restrict__ st) {
    if (st->converged) return;
    int idx = blockIdx.x * blockDim.x + threadIdx.x;
    Z[idx] = ZN[idx];
}

__global__ void state_update_kernel(SolverState* st) {
    if (st->converged) return;
    float e = __uint_as_float(st->err_bits);
    if (e > st->prev_err) st->alpha = fmaxf(st->alpha * 0.1f, 1e-6f);
    st->prev_err = e;
    if (!(e > TOLF) || __builtin_isnan(e)) st->converged = 1;  // exit on err<=TOL or NaN
    st->err_bits = 0u;
}

// Hot kernel: S = T @ A  (T: 128x8192 bf16, A: 8192x8192 f32 -> bf16 in LDS)
// 4-buffer LDS pipeline with parity-indexed register stages (no rotation copies),
// one LDS-only barrier per K-step, branch-free steady state via clamped indices.
// mode 0: ZN = relu(z + alpha*(S + bias - z)); err = max|ZN - z|  (guarded)
// mode 1: OUT = relu(S + bias)                                    (final, unguarded)
__global__ __launch_bounds__(256)
void zA_wmma_kernel(const unsigned short* __restrict__ T,
                    const float* __restrict__ Agraph,
                    const float* __restrict__ bias,
                    const float* __restrict__ Z,
                    float* __restrict__ OUT,
                    SolverState* __restrict__ st, int mode) {
    if (mode == 0 && st->converged) return;

    const int tid  = threadIdx.x;
    const int wave = tid >> 5;
    const int lane = tid & 31;
    const int n0   = blockIdx.x * NB;
    const float alpha = st->alpha;

    __shared__ unsigned short Abt[4][NB * LDS_STRIDE];  // 4 rotating bf16 tiles, 10 KB
    __shared__ unsigned err_sh;
    if (tid == 0) err_sh = 0u;

    // all 256 threads stage the 32x32 tile: 2 rows x 2 cols each
    const int r_ld = (tid >> 4) * 2;   // 0,2,..,30   (k pair within tile)
    const int c_ld = (tid & 15) * 2;   // 0,2,..,30   (col pair)

    // prologue: tiles 0 and 1 into Abt[0]/Abt[1]
    {
        float2 q0, q1;
        load_tile(Agraph, 0, n0, r_ld, c_ld, q0, q1);
        store_tile(&Abt[0][0], c_ld, r_ld, q0, q1);
        load_tile(Agraph, 1, n0, r_ld, c_ld, q0, q1);
        store_tile(&Abt[1][0], c_ld, r_ld, q0, q1);
    }
    __syncthreads();

    v8f acc[2] = {};   // 2 N-tiles of 16, this wave's 16-row strip

    // per-lane A-operand (T) base: lanes 0-15 -> M=lane, chunks {k0..k0+7, k0+16..23}
    //                              lanes 16-31 -> M=lane-16, chunks {k0+8..15, k0+24..31}
    const unsigned short* Trow = T + (size_t)(wave * 16 + (lane & 15)) * N_DIM
                                   + ((lane < 16) ? 0 : 8);
    const int b_koff = (lane < 16) ? 0 : 8;        // B-operand K half within tile
    const int b_elem = (lane & 15) * LDS_STRIDE + b_koff;

    // parity-indexed pipeline registers (indices constant after unroll-by-4)
    v16bf aF[2], b0F[2], b1F[2];
    float2 pA[2], pB[2];

    aF[0]  = load_afrag(Trow);
    b0F[0] = load_bfrag(&Abt[0][b_elem]);
    b1F[0] = load_bfrag(&Abt[0][b_elem + 16 * LDS_STRIDE]);
    load_tile(Agraph, 2, n0, r_ld, c_ld, pA[0], pB[0]);   // tile 2 in flight

#pragma unroll 4
    for (int s = 0; s < NSTEPS; ++s) {
        const int cur  = s & 1;
        const int nxt  = cur ^ 1;
        const int nbuf = (s + 1) & 3;   // buffer holding tile s+1
        const int sbuf = (s + 2) & 3;   // buffer to receive tile s+2

        // clamped pipeline indices: tail reads hit valid memory but feed
        // fragments that are never consumed -> branch-free body.
        const int s1 = (s + 1 < NSTEPS) ? s + 1 : NSTEPS - 1;
        const int s3 = (s + 3 < NSTEPS) ? s + 3 : NSTEPS - 1;
        const int s4 = (s + 4 < NSTEPS) ? s + 4 : NSTEPS - 1;

        // preload next step's fragments (consumed after the barrier below)
        aF[nxt]  = load_afrag(Trow + s1 * KB);
        b0F[nxt] = load_bfrag(&Abt[nbuf][b_elem]);
        b1F[nxt] = load_bfrag(&Abt[nbuf][b_elem + 16 * LDS_STRIDE]);

        // global load tile s+3; L2 prefetch tile s+4
        load_tile(Agraph, s3, n0, r_ld, c_ld, pA[nxt], pB[nxt]);
        __builtin_prefetch(Agraph + (size_t)(s4 * KB + r_ld) * N_DIM + (n0 + c_ld), 0, 1);

        // compute tile s (fragments were loaded a full step ago -> no wait stall)
        acc[0] = __builtin_amdgcn_wmma_f32_16x16x32_bf16(
                     false, aF[cur], false, b0F[cur], (short)0, acc[0], false, false);
        acc[1] = __builtin_amdgcn_wmma_f32_16x16x32_bf16(
                     false, aF[cur], false, b1F[cur], (short)0, acc[1], false, false);

        // stage tile s+2 (globally loaded one step ago) into its buffer
        store_tile(&Abt[sbuf][0], c_ld, r_ld, pA[cur], pB[cur]);

        // LDS-only barrier: keeps global loads in flight across steps
        wg_barrier_lds();
    }

    // epilogue: C/D layout -> lanes 0-15: M=r, N=lane; lanes 16-31: M=r+8, N=lane-16
    unsigned localmax = 0u;
#pragma unroll
    for (int nt = 0; nt < 2; ++nt) {
#pragma unroll
        for (int r = 0; r < 8; ++r) {
            const int m = wave * 16 + r + ((lane < 16) ? 0 : 8);
            const int n = n0 + nt * 16 + (lane & 15);
            const size_t off = (size_t)m * N_DIM + n;
            const float s = acc[nt][r] + bias[off];
            if (mode == 0) {
                const float zv = Z[off];
                const float zn = fmaxf(fmaf(alpha, s - zv, zv), 0.f);  // (1-a)z + a*s
                OUT[off] = zn;
                localmax = max(localmax, __float_as_uint(fabsf(zn - zv)));
            } else {
                OUT[off] = fmaxf(s, 0.f);
            }
        }
    }
    if (mode == 0) {
        atomicMax(&err_sh, localmax);
        __syncthreads();
        if (tid == 0) atomicMax(&st->err_bits, err_sh);
    }
}

extern "C" void kernel_launch(void* const* d_in, const int* in_sizes, int n_in,
                              void* d_out, int out_size, void* d_ws, size_t ws_size,
                              hipStream_t stream) {
    const float* x = (const float*)d_in[0];   // [P, N]
    const float* W = (const float*)d_in[1];   // [D, D]
    const float* U = (const float*)d_in[2];   // [D, P]
    const float* A = (const float*)d_in[3];   // [N, N]

    const size_t DN  = (size_t)D_DIM * N_DIM;          // 1,048,576
    const size_t F32 = DN * sizeof(float);             // 4 MB
    char* ws = (char*)d_ws;
    float*          bias = (float*)(ws);               // 4 MB
    float*          Z    = (float*)(ws + F32);         // 4 MB
    float*          ZN   = (float*)(ws + 2 * F32);     // 4 MB
    unsigned short* T    = (unsigned short*)(ws + 3 * F32);          // 2 MB
    SolverState*    st   = (SolverState*)(ws + 3 * F32 + DN * 2);

    const int  G = (int)(DN / 256);            // 4096 blocks of 256
    const dim3 gemmGrid(N_DIM / 256, D_DIM);   // (32, 128)

    init_kernel<<<G, 256, 0, stream>>>(Z, st);
    // bias = U @ x  (unguarded, f32 out)
    small_gemm_kernel<<<gemmGrid, 256, 0, stream>>>(U, x, bias, nullptr, st, 0);

    for (int it = 0; it < MAX_ITER; ++it) {
        // T = bf16(W @ Z)                     (guarded)
        small_gemm_kernel<<<gemmGrid, 256, 0, stream>>>(W, Z, nullptr, T, st, 1);
        // ZN = relu((1-a)Z + a*(T@A + bias)); err                   (guarded)
        zA_wmma_kernel<<<N_DIM / NB, 256, 0, stream>>>(T, A, bias, Z, ZN, st, 0);
        // Z = ZN                              (guarded)
        commit_kernel<<<G, 256, 0, stream>>>(Z, ZN, st);
        // alpha / prev_err / converged update
        state_update_kernel<<<1, 1, 0, stream>>>(st);
    }

    // final: out = relu(W @ Z @ A + bias)  (unguarded)
    small_gemm_kernel<<<gemmGrid, 256, 0, stream>>>(W, Z, nullptr, T, st, 0);
    zA_wmma_kernel<<<N_DIM / NB, 256, 0, stream>>>(T, A, bias, nullptr,
                                                   (float*)d_out, st, 1);
}
